// SpectralConv2d_31774168056089
// MI455X (gfx1250) — matrix operand order repaired and checked by reading, not compiled, not run on hardware
//
#include <hip/hip_runtime.h>
#include <hip/hip_bf16.h>
#include <math.h>

typedef __attribute__((ext_vector_type(16))) __bf16 v16bf;
typedef __attribute__((ext_vector_type(8)))  __bf16 v8bf;
typedef __attribute__((ext_vector_type(8)))  float  v8f;

#define PI_F 3.14159265358979323846f

// ---------------------------------------------------------------------------
// WMMA fragment helpers (wave32, ISA 7.12.2 layouts), vectorized.
// A matrix 16x32 bf16: lane<16 -> row=lane, K in {0..7,16..23};
//                      lane>=16 -> row=lane-16, K in {8..15,24..31}
// -> two 16-byte contiguous loads per lane.
// ---------------------------------------------------------------------------
__device__ inline v16bf load_a_frag(const __bf16* tile, int ld) {
  int lane  = threadIdx.x & 31;
  int row   = lane & 15;
  int khalf = (lane >> 4) * 8;
  const __bf16* p = tile + row * ld + khalf;
  v8bf lo = *(const v8bf*)p;          // K = khalf .. khalf+7
  v8bf hi = *(const v8bf*)(p + 16);   // K = khalf+16 .. khalf+23
  return __builtin_shufflevector(lo, hi, 0, 1, 2, 3, 4, 5, 6, 7,
                                 8, 9, 10, 11, 12, 13, 14, 15);
}

// B matrix 32x16 bf16: N = lane&15, K in {0..7,16..23} / {8..15,24..31} like A.
__device__ inline v16bf load_b_frag16(const __bf16* p, int ld) {
  int lane = threadIdx.x & 31;
  int n    = lane & 15;
  int kh   = (lane >> 4) * 8;
  v16bf r;
#pragma unroll
  for (int s = 0; s < 16; ++s) r[s] = p[(size_t)(kh + (s & 7) + (s >> 3) * 16) * ld + n];
  return r;
}

// ---------------------------------------------------------------------------
// Misc small kernels
// ---------------------------------------------------------------------------
__global__ void k_cvt_bf16(const float* __restrict__ s, __bf16* __restrict__ d, int n) {
  for (int i = blockIdx.x * blockDim.x + threadIdx.x; i < n; i += gridDim.x * blockDim.x)
    d[i] = (__bf16)s[i];
}

// Spectral weight transpose: w(I,O,16,16,2) f32 -> wTr/wTi[mode][i][o] bf16,
// mode = kxIdx*16+ky (kxIdx<16 -> fw1, else fw2). Coalesced writes.
__global__ void k_wprep(const float* __restrict__ fw1, const float* __restrict__ fw2,
                        __bf16* __restrict__ wTr, __bf16* __restrict__ wTi) {
  const int total = 512 * 64 * 64;
  for (int idx = blockIdx.x * blockDim.x + threadIdx.x; idx < total;
       idx += gridDim.x * blockDim.x) {
    int o = idx & 63;
    int i = (idx >> 6) & 63;
    int mode = idx >> 12;
    int ky = mode & 15;
    int kxIdx = mode >> 4;
    const float* w = (kxIdx < 16) ? fw1 : fw2;
    int wkx = kxIdx & 15;
    size_t widx = ((((size_t)i * 64 + o) * 16 + wkx) * 16 + ky) * 2;
    wTr[idx] = (__bf16)w[widx];
    wTi[idx] = (__bf16)w[widx + 1];
  }
}

// xm[b][i][m][n] = x[b][(m*128+n)*64+i] - z[b][0][m][n][i]
__global__ void k_prep(const float* __restrict__ x, const float* __restrict__ z,
                       float* __restrict__ xm) {
  const size_t total = (size_t)4 * 64 * 128 * 128;
  for (size_t idx = (size_t)blockIdx.x * blockDim.x + threadIdx.x; idx < total;
       idx += (size_t)gridDim.x * blockDim.x) {
    int n = idx & 127;
    int m = (idx >> 7) & 127;
    int i = (idx >> 14) & 63;
    int b = (int)(idx >> 20);
    size_t sp = (size_t)m * 128 + n;
    float xv = x[((size_t)b * 16384 + sp) * 64 + i];
    float zv = z[(((size_t)b * 2 + 0) * 16384 + sp) * 64 + i];
    xm[idx] = xv - zv;
  }
}

// ---------------------------------------------------------------------------
// Forward DFT stage 1 (over n, keep ky=0..15).  Row = (b*64+i)*128+m.
// ---------------------------------------------------------------------------
__global__ __launch_bounds__(128) void k_dft_cols(const float* __restrict__ xm,
                                                  float* __restrict__ Ar,
                                                  float* __restrict__ Ai) {
  __shared__ float xrow[4 * 128];
  __shared__ float ct[128], st[128];
  const int tid = threadIdx.x;
  const int rowbase = blockIdx.x * 4;
  if (tid < 128) {
    float ang = (float)tid * (2.0f * PI_F / 128.0f);
    __sincosf(ang, &st[tid], &ct[tid]);
  }
  for (int e = tid; e < 512; e += 128)
    xrow[e] = xm[(size_t)(rowbase + (e >> 7)) * 128 + (e & 127)];
  __syncthreads();

  int lr   = tid >> 5;
  int q    = tid & 31;
  int ky   = q >> 1;
  int comp = q & 1;
  float acc = 0.0f;
#pragma unroll 4
  for (int n = 0; n < 128; ++n) {
    int ti = (ky * n) & 127;
    float xv = xrow[lr * 128 + n];
    acc += comp ? (-xv * st[ti]) : (xv * ct[ti]);
  }
  size_t o = (size_t)(rowbase + lr) * 16 + ky;
  if (comp) Ai[o] = acc; else Ar[o] = acc;
}

// ---------------------------------------------------------------------------
// Forward DFT stage 2 (over m, keep kx in {0..15,112..127}), ortho 1/128.
// Z[b][i][kxIdx][ky], one block per (b,i)
// ---------------------------------------------------------------------------
__global__ __launch_bounds__(256) void k_dft_rows(const float* __restrict__ Ar,
                                                  const float* __restrict__ Ai,
                                                  float* __restrict__ Zr,
                                                  float* __restrict__ Zi) {
  __shared__ float sAr[128 * 16], sAi[128 * 16];
  __shared__ float ct[128], st[128];
  const int tid = threadIdx.x;
  const size_t base = (size_t)blockIdx.x * 2048;
  if (tid < 128) {
    float ang = (float)tid * (2.0f * PI_F / 128.0f);
    __sincosf(ang, &st[tid], &ct[tid]);
  }
  for (int e = tid; e < 2048; e += 256) { sAr[e] = Ar[base + e]; sAi[e] = Ai[base + e]; }
  __syncthreads();

#pragma unroll
  for (int qq = 0; qq < 4; ++qq) {
    int oid   = tid + 256 * qq;
    int comp  = oid & 1;
    int ky    = (oid >> 1) & 15;
    int kxIdx = oid >> 5;
    int kx    = (kxIdx < 16) ? kxIdx : (96 + kxIdx);
    float acc = 0.0f;
    for (int m = 0; m < 128; ++m) {
      int ti = (kx * m) & 127;
      float ar = sAr[m * 16 + ky], ai = sAi[m * 16 + ky];
      acc += comp ? (ai * ct[ti] - ar * st[ti]) : (ar * ct[ti] + ai * st[ti]);
    }
    acc *= (1.0f / 128.0f);
    size_t o = ((size_t)blockIdx.x * 32 + kxIdx) * 16 + ky;
    if (comp) Zi[o] = acc; else Zr[o] = acc;
  }
}

// ---------------------------------------------------------------------------
// Mode mixing: per mode, complex Out[b,o] = sum_i Z[b,i]*W[i,o].
// Z staged into zero-padded 16x64 LDS tiles (no divergent guarded loads);
// weights read from pre-transposed mode-major bf16 (consecutive lanes read consecutive o).
// ---------------------------------------------------------------------------
__global__ __launch_bounds__(128) void k_mix(const float* __restrict__ Zr,
                                             const float* __restrict__ Zi,
                                             const __bf16* __restrict__ wTr,
                                             const __bf16* __restrict__ wTi,
                                             float* __restrict__ Or_,
                                             float* __restrict__ Oi_) {
  __shared__ __bf16 sZr[16 * 64], sZi[16 * 64];
  const int mode  = blockIdx.x;
  const int ky    = mode & 15;
  const int kxIdx = mode >> 4;
  const int tid   = threadIdx.x;

  for (int e = tid; e < 1024; e += 128) { sZr[e] = (__bf16)0.0f; sZi[e] = (__bf16)0.0f; }
  __syncthreads();
  for (int e = tid; e < 256; e += 128) {       // b = 0..3, i = 0..63
    int b = e >> 6, i = e & 63;
    size_t zidx = (((size_t)b * 64 + i) * 32 + kxIdx) * 16 + ky;
    sZr[b * 64 + i] = (__bf16)Zr[zidx];
    sZi[b * 64 + i] = (__bf16)Zi[zidx];
  }
  __syncthreads();

  const int lane = tid & 31;
  const int wave = tid >> 5;                   // o-tile
  const __bf16* wr_base = wTr + (size_t)mode * 4096;
  const __bf16* wi_base = wTi + (size_t)mode * 4096;

  v8f ar_acc = {}, ai_acc = {};
#pragma unroll
  for (int k0 = 0; k0 < 64; k0 += 32) {
    v16bf azr = load_a_frag(sZr + k0, 64);
    v16bf azi = load_a_frag(sZi + k0, 64);
    int ki = k0 + lane;
    v16bf bwr = load_b_frag16(wr_base + (size_t)k0 * 64 + wave * 16, 64);
    v16bf bwi = load_b_frag16(wi_base + (size_t)k0 * 64 + wave * 16, 64);
    v16bf bwin;
#pragma unroll
    for (int j = 0; j < 16; ++j) bwin[j] = (__bf16)(-(float)bwi[j]);
    ar_acc = __builtin_amdgcn_wmma_f32_16x16x32_bf16(false, azr, false, bwr,  (short)0, ar_acc, false, false);
    ar_acc = __builtin_amdgcn_wmma_f32_16x16x32_bf16(false, azi, false, bwin, (short)0, ar_acc, false, false);
    ai_acc = __builtin_amdgcn_wmma_f32_16x16x32_bf16(false, azr, false, bwi,  (short)0, ai_acc, false, false);
    ai_acc = __builtin_amdgcn_wmma_f32_16x16x32_bf16(false, azi, false, bwr,  (short)0, ai_acc, false, false);
  }
  int bb   = (lane >> 4) * 8;
  int ocol = wave * 16 + (lane & 15);
#pragma unroll
  for (int r = 0; r < 8; ++r) {
    int b = bb + r;
    if (b < 4) {
      size_t oidx = (((size_t)b * 64 + ocol) * 32 + kxIdx) * 16 + ky;
      Or_[oidx] = ar_acc[r];
      Oi_[oidx] = ai_acc[r];
    }
  }
}

// ---------------------------------------------------------------------------
// Inverse DFT over rows: T[b][o][m][ky] = sum_kx O * e^{+2pi i kx m/128}
// ---------------------------------------------------------------------------
__global__ __launch_bounds__(256) void k_idft_rows(const float* __restrict__ Or_,
                                                   const float* __restrict__ Oi_,
                                                   float* __restrict__ Tr,
                                                   float* __restrict__ Ti) {
  __shared__ float sOr[32 * 16], sOi[32 * 16];
  __shared__ float ct[128], st[128];
  const int tid = threadIdx.x;
  const size_t base = (size_t)blockIdx.x * 512;
  if (tid < 128) {
    float ang = (float)tid * (2.0f * PI_F / 128.0f);
    __sincosf(ang, &st[tid], &ct[tid]);
  }
  for (int e = tid; e < 512; e += 256) { sOr[e] = Or_[base + e]; sOi[e] = Oi_[base + e]; }
  __syncthreads();

#pragma unroll
  for (int qq = 0; qq < 16; ++qq) {
    int oid  = tid + 256 * qq;
    int comp = oid & 1;
    int ky   = (oid >> 1) & 15;
    int m    = oid >> 5;
    float acc = 0.0f;
    for (int kxIdx = 0; kxIdx < 32; ++kxIdx) {
      int kx = (kxIdx < 16) ? kxIdx : (96 + kxIdx);
      int ti = (kx * m) & 127;
      float orv = sOr[kxIdx * 16 + ky], oiv = sOi[kxIdx * 16 + ky];
      acc += comp ? (orv * st[ti] + oiv * ct[ti]) : (orv * ct[ti] - oiv * st[ti]);
    }
    size_t o = ((size_t)blockIdx.x * 128 + m) * 16 + ky;
    if (comp) Ti[o] = acc; else Tr[o] = acc;
  }
}

// ---------------------------------------------------------------------------
// Inverse DFT over cols, real output, Hermitian weights, ortho 1/128.
// ---------------------------------------------------------------------------
__global__ __launch_bounds__(256) void k_idft_cols(const float* __restrict__ Tr,
                                                   const float* __restrict__ Ti,
                                                   float* __restrict__ xo) {
  __shared__ float ct[128], st[128];
  const int tid = threadIdx.x;
  if (tid < 128) {
    float ang = (float)tid * (2.0f * PI_F / 128.0f);
    __sincosf(ang, &st[tid], &ct[tid]);
  }
  __syncthreads();
  const size_t total = (size_t)4 * 128 * 128 * 64;
  for (size_t idx = (size_t)blockIdx.x * blockDim.x + tid; idx < total;
       idx += (size_t)gridDim.x * blockDim.x) {
    int o = idx & 63;
    int n = (idx >> 6) & 127;
    int m = (idx >> 13) & 127;
    int b = (int)(idx >> 20);
    size_t tbase = (((size_t)b * 64 + o) * 128 + m) * 16;
    float acc = 0.0f;
#pragma unroll
    for (int ky = 0; ky < 16; ++ky) {
      int ti = (ky * n) & 127;
      float alpha = (ky == 0) ? 1.0f : 2.0f;
      acc += alpha * (Tr[tbase + ky] * ct[ti] - Ti[tbase + ky] * st[ti]);
    }
    xo[idx] = acc * (1.0f / 128.0f);
  }
}

// ---------------------------------------------------------------------------
// Tiled WMMA GEMM: out = A(Mrows x K) * B(K x N, bf16) + bias
// 64x64 tile/block, BK=32, 8 waves: wave -> M-subtile (w&3), N-group (w>>2)
// ---------------------------------------------------------------------------
template <typename TA, bool OUT_BF16>
__global__ __launch_bounds__(256) void k_gemm(const TA* __restrict__ A,
                                              const __bf16* __restrict__ B,
                                              const float* __restrict__ bias,
                                              void* __restrict__ out,
                                              int Mrows, int K, int N) {
  __shared__ __bf16 sA[64 * 32];
  __shared__ __bf16 sB[32 * 64];
  const int tid  = threadIdx.x;
  const int lane = tid & 31;
  const int wave = tid >> 5;
  const int row0 = blockIdx.x * 64;
  const int col0 = blockIdx.y * 64;
  const int mt   = wave & 3;
  const int ng   = wave >> 2;

  v8f acc0 = {}, acc1 = {};
  for (int kt = 0; kt < K; kt += 32) {
    if (kt + 32 < K)
      __builtin_prefetch(&B[(size_t)(kt + 32 + (tid >> 3)) * N + col0 + ((tid & 7) << 3)], 0, 1);
#pragma unroll
    for (int j = 0; j < 8; ++j) {
      int e = tid * 8 + j;
      sA[e] = (__bf16)(float)A[(size_t)(row0 + (e >> 5)) * K + kt + (e & 31)];
    }
#pragma unroll
    for (int j = 0; j < 8; ++j) {
      int e = tid * 8 + j;
      sB[e] = B[(size_t)(kt + (e >> 6)) * N + col0 + (e & 63)];
    }
    __syncthreads();
    v16bf a  = load_a_frag(sA + mt * 16 * 32, 32);
    v16bf b0 = load_b_frag16(sB + ng * 32, 64);
    v16bf b1 = load_b_frag16(sB + ng * 32 + 16, 64);
    acc0 = __builtin_amdgcn_wmma_f32_16x16x32_bf16(false, a, false, b0, (short)0, acc0, false, false);
    acc1 = __builtin_amdgcn_wmma_f32_16x16x32_bf16(false, a, false, b1, (short)0, acc1, false, false);
    __syncthreads();
  }

  int crow  = row0 + mt * 16 + (lane >> 4) * 8;
  int ccol0 = col0 + ng * 32 + (lane & 15);
  float bv0 = bias[ccol0];
  float bv1 = bias[ccol0 + 16];
#pragma unroll
  for (int r = 0; r < 8; ++r) {
    float v0 = acc0[r] + bv0;
    float v1 = acc1[r] + bv1;
    size_t i0 = (size_t)(crow + r) * N + ccol0;
    if (OUT_BF16) {
      __bf16* op = (__bf16*)out;
      op[i0] = (__bf16)v0;
      op[i0 + 16] = (__bf16)v1;
    } else {
      float* op = (float*)out;
      op[i0] = v0;
      op[i0 + 16] = v1;
    }
  }
}

// ---------------------------------------------------------------------------
// GroupNorm statistics: block per (b, group); deterministic tree reduction.
// ---------------------------------------------------------------------------
template <typename T>
__global__ __launch_bounds__(256) void k_gn_stats(const T* __restrict__ x,
                                                  float* __restrict__ stats, int C) {
  __shared__ float ssum[256], ssq[256];
  const int groups = C >> 4;
  const int b  = blockIdx.x / groups;
  const int gi = blockIdx.x % groups;
  const int tid = threadIdx.x;
  float s = 0.0f, q = 0.0f;
  for (int e = tid; e < 16384 * 16; e += 256) {
    int sp = e >> 4, c = e & 15;
    float v = (float)x[((size_t)b * 16384 + sp) * C + gi * 16 + c];
    s += v;
    q += v * v;
  }
  ssum[tid] = s; ssq[tid] = q;
  __syncthreads();
  for (int off = 128; off > 0; off >>= 1) {
    if (tid < off) { ssum[tid] += ssum[tid + off]; ssq[tid] += ssq[tid + off]; }
    __syncthreads();
  }
  if (tid == 0) {
    const float inv = 1.0f / (16384.0f * 16.0f);
    float mean = ssum[0] * inv;
    float var  = ssq[0] * inv - mean * mean;
    stats[2 * blockIdx.x]     = mean;
    stats[2 * blockIdx.x + 1] = rsqrtf(var + 1e-5f);
  }
}

__global__ void k_gn_relu_bf16(__bf16* __restrict__ h, const float* __restrict__ gw,
                               const float* __restrict__ gb, const float* __restrict__ stats) {
  const size_t total = (size_t)4 * 16384 * 256;
  for (size_t idx = (size_t)blockIdx.x * blockDim.x + threadIdx.x; idx < total;
       idx += (size_t)gridDim.x * blockDim.x) {
    int c = idx & 255;
    int b = (int)(idx >> 22);
    int gi = c >> 4;
    float mean = stats[2 * (b * 16 + gi)];
    float rstd = stats[2 * (b * 16 + gi) + 1];
    float v = ((float)h[idx] - mean) * rstd * gw[c] + gb[c];
    h[idx] = (__bf16)fmaxf(v, 0.0f);
  }
}

__global__ void k_gn_res(const float* __restrict__ h, const float* __restrict__ z, int path,
                         const float* __restrict__ gw, const float* __restrict__ gb,
                         const float* __restrict__ stats, float* __restrict__ t) {
  const size_t total = (size_t)4 * 16384 * 64;
  for (size_t idx = (size_t)blockIdx.x * blockDim.x + threadIdx.x; idx < total;
       idx += (size_t)gridDim.x * blockDim.x) {
    int c = idx & 63;
    size_t b = idx >> 20;
    int gi = c >> 4;
    float mean = stats[2 * (b * 4 + gi)];
    float rstd = stats[2 * (b * 4 + gi) + 1];
    float v = (h[idx] - mean) * rstd * gw[c] + gb[c];
    size_t local = idx - (b << 20);
    t[idx] = z[((b * 2 + path) << 20) + local] + v;
  }
}

__global__ void k_gn_out(const float* __restrict__ t, const float* __restrict__ gw,
                         const float* __restrict__ gb, const float* __restrict__ stats,
                         float* __restrict__ out, int path) {
  const size_t total = (size_t)4 * 16384 * 64;
  for (size_t idx = (size_t)blockIdx.x * blockDim.x + threadIdx.x; idx < total;
       idx += (size_t)gridDim.x * blockDim.x) {
    int c = idx & 63;
    size_t b = idx >> 20;
    int gi = c >> 4;
    float mean = stats[2 * (b * 4 + gi)];
    float rstd = stats[2 * (b * 4 + gi) + 1];
    float v = (t[idx] - mean) * rstd * gw[c] + gb[c];
    size_t local = idx - (b << 20);
    out[((b * 2 + path) << 20) + local] = v;
  }
}

// ---------------------------------------------------------------------------
// Host side
// ---------------------------------------------------------------------------
static void run_ff(const float* xo, const __bf16* w1, const float* b1,
                   const __bf16* w2, const float* b2,
                   const float* g1w, const float* g1b,
                   const float* g2w, const float* g2b,
                   const float* g3w, const float* g3b,
                   const float* z, int path,
                   __bf16* h1, float* h2, float* t, float* stats,
                   float* out, hipStream_t stream) {
  dim3 gg1(1024, 4);
  k_gemm<float, true><<<gg1, 256, 0, stream>>>(xo, w1, b1, (void*)h1, 65536, 64, 256);
  k_gn_stats<__bf16><<<64, 256, 0, stream>>>(h1, stats, 256);
  k_gn_relu_bf16<<<2048, 256, 0, stream>>>(h1, g1w, g1b, stats);
  dim3 gg2(1024, 1);
  k_gemm<__bf16, false><<<gg2, 256, 0, stream>>>(h1, w2, b2, (void*)h2, 65536, 256, 64);
  k_gn_stats<float><<<16, 256, 0, stream>>>(h2, stats, 64);
  k_gn_res<<<2048, 256, 0, stream>>>(h2, z, path, g2w, g2b, stats, t);
  k_gn_stats<float><<<16, 256, 0, stream>>>(t, stats, 64);
  k_gn_out<<<2048, 256, 0, stream>>>(t, g3w, g3b, stats, out, path);
}

extern "C" void kernel_launch(void* const* d_in, const int* in_sizes, int n_in,
                              void* d_out, int out_size, void* d_ws, size_t ws_size,
                              hipStream_t stream) {
  const float* z   = (const float*)d_in[0];
  const float* x   = (const float*)d_in[1];
  const float* fw1 = (const float*)d_in[2];
  const float* fw2 = (const float*)d_in[3];
  const float* b_l1w = (const float*)d_in[4];
  const float* b_l1b = (const float*)d_in[5];
  const float* b_l2w = (const float*)d_in[6];
  const float* b_l2b = (const float*)d_in[7];
  const float* b_g1w = (const float*)d_in[8];
  const float* b_g1b = (const float*)d_in[9];
  const float* b_g2w = (const float*)d_in[10];
  const float* b_g2b = (const float*)d_in[11];
  const float* b_g3w = (const float*)d_in[12];
  const float* b_g3b = (const float*)d_in[13];
  const float* f_l1w = (const float*)d_in[14];
  const float* f_l1b = (const float*)d_in[15];
  const float* f_l2w = (const float*)d_in[16];
  const float* f_l2b = (const float*)d_in[17];
  const float* f_g1w = (const float*)d_in[18];
  const float* f_g1b = (const float*)d_in[19];
  const float* f_g2w = (const float*)d_in[20];
  const float* f_g2b = (const float*)d_in[21];
  const float* f_g3w = (const float*)d_in[22];
  const float* f_g3b = (const float*)d_in[23];
  float* out = (float*)d_out;

  char* w = (char*)d_ws;
  float* xo = (float*)w;                       // 16 MB, persistent
  char*  S  = w + 16777216;                    // 64 MB reused region
  // phase-1 view of S
  float* xm  = (float*)S;                      // 16 MB
  float* Ar  = (float*)(S + 16777216);         // 2 MB
  float* Ai  = (float*)(S + 18874368);         // 2 MB
  float* Zr  = (float*)(S + 20971520);         // 0.5 MB
  float* Zi  = (float*)(S + 21495808);
  float* Or_ = (float*)(S + 22020096);
  float* Oi_ = (float*)(S + 22544384);
  float* Tr  = (float*)(S + 23068672);         // 2 MB
  float* Ti  = (float*)(S + 25165824);         // 2 MB
  // FF view of S
  __bf16* h1 = (__bf16*)S;                     // 32 MB
  float*  h2 = (float*)(S + 33554432);         // 16 MB
  float*  t  = (float*)(S + 50331648);         // 16 MB
  char* W = w + 16777216 + 67108864;           // weight/stat region
  __bf16* bw1  = (__bf16*)W;
  __bf16* bw2  = (__bf16*)(W + 32768);
  __bf16* fw1b = (__bf16*)(W + 65536);
  __bf16* fw2b = (__bf16*)(W + 98304);
  float*  stats = (float*)(W + 131072);        // 512 B
  __bf16* wTr = (__bf16*)(W + 262144);         // 4 MB
  __bf16* wTi = (__bf16*)(W + 262144 + 4194304);

  // weight conversion / transposition
  k_cvt_bf16<<<64, 256, 0, stream>>>(b_l1w, bw1, 64 * 256);
  k_cvt_bf16<<<64, 256, 0, stream>>>(b_l2w, bw2, 256 * 64);
  k_cvt_bf16<<<64, 256, 0, stream>>>(f_l1w, fw1b, 64 * 256);
  k_cvt_bf16<<<64, 256, 0, stream>>>(f_l2w, fw2b, 256 * 64);
  k_wprep<<<4096, 256, 0, stream>>>(fw1, fw2, wTr, wTi);

  // spectral pipeline
  k_prep<<<4096, 256, 0, stream>>>(x, z, xm);
  k_dft_cols<<<8192, 128, 0, stream>>>(xm, Ar, Ai);
  k_dft_rows<<<256, 256, 0, stream>>>(Ar, Ai, Zr, Zi);
  k_mix<<<512, 128, 0, stream>>>(Zr, Zi, wTr, wTi, Or_, Oi_);
  k_idft_rows<<<256, 256, 0, stream>>>(Or_, Oi_, Tr, Ti);
  k_idft_cols<<<4096, 256, 0, stream>>>(Tr, Ti, xo);

  // feed-forward: backcast (path 0), forecast (path 1)
  run_ff(xo, bw1, b_l1b, bw2, b_l2b, b_g1w, b_g1b, b_g2w, b_g2b, b_g3w, b_g3b,
         z, 0, h1, h2, t, stats, out, stream);
  run_ff(xo, fw1b, f_l1b, fw2b, f_l2b, f_g1w, f_g1b, f_g2w, f_g2b, f_g3w, f_g3b,
         z, 1, h1, h2, t, stats, out, stream);
}
